// MultiBoxLoss_6390911336616
// MI455X (gfx1250) — compile-verified, matrix operand order checked by reading, not run on hardware
//
#include <hip/hip_runtime.h>
#include <hip/hip_bf16.h>
#include <stdint.h>

typedef __attribute__((ext_vector_type(16))) _Float16 v16h;
typedef __attribute__((ext_vector_type(8)))  float    v8f;
typedef __attribute__((ext_vector_type(4)))  int      v4i_;

#define NB 32
#define NA 16384
#define NC 81

// numerically stable BCE-with-logits: softplus(x) - t*x
__device__ __forceinline__ float bce_logits(float x, float t) {
    return fmaxf(x, 0.0f) + log1pf(expf(-fabsf(x))) - t * x;
}

// ---------------------------------------------------------------------------
// Pass 1: per-anchor max over C of ce, zeroed where depth != 0.
// One wave (32 lanes) per anchor row; lane-strided coalesced loads.
// ---------------------------------------------------------------------------
__global__ void k_rowmax(const float* __restrict__ pred,
                         const float* __restrict__ tgt,
                         const int*   __restrict__ depth,
                         float*       __restrict__ maxce) {
    const int wavesPerBlk = blockDim.x >> 5;
    const int row  = blockIdx.x * wavesPerBlk + (threadIdx.x >> 5);
    const int lane = threadIdx.x & 31;
    if (row >= NB * NA) return;
    const size_t base = (size_t)row * NC;
    float m = -__builtin_inff();
    for (int c = lane; c < NC; c += 32) {
        float x = pred[base + c];
        float t = tgt[base + c];
        m = fmaxf(m, bce_logits(x, t));
    }
    #pragma unroll
    for (int off = 16; off > 0; off >>= 1)
        m = fmaxf(m, __shfl_xor(m, off, 32));
    if (lane == 0)
        maxce[row] = (depth[row] != 0) ? 0.0f : m;
}

// ---------------------------------------------------------------------------
// Pass 2: one workgroup per batch.
//   (a) num_pos via WMMA reduction: 16x32 f16 indicator tile x ones(32x16),
//       D row-sums -> chunk counts on the matrix pipe (exact: 0/1 in f16,
//       f32 accumulate).
//   (b) exact rank: 16384 x 64-bit keys sorted ascending in 128KB of LDS
//       (CDNA5 320KB/WGP). key = (~orderedBits(max_ce) << 32) | index, so
//       ascending sort == stable argsort(-max_ce); sorted position == rank.
//   (c) scatter neg flag = (rank < num_neg).
// ---------------------------------------------------------------------------
__global__ __launch_bounds__(1024)
void k_rank(const int*   __restrict__ depth,
            const int*   __restrict__ ratio_p,
            const float* __restrict__ maxce,
            unsigned char* __restrict__ negmask) {
    extern __shared__ unsigned char smem[];
    uint64_t* s_key    = (uint64_t*)smem;                    // 16384 * 8 B
    float*    s_wsum   = (float*)(smem + (size_t)NA * 8);    // 32 floats
    int*      s_numneg = (int*)(smem + (size_t)NA * 8 + 128);

    const int b    = blockIdx.x;
    const int tid  = threadIdx.x;
    const int nthr = blockDim.x;
    const int wave = tid >> 5;
    const int lane = tid & 31;

    // ---- (a) num_pos on the matrix pipe -------------------------------
    {
        const size_t chunk = (size_t)b * NA + (size_t)wave * 512 + (size_t)lane * 16;
        const v4i_* dp = (const v4i_*)(depth + chunk);      // 64B aligned
        v16h a;
        #pragma unroll
        for (int q = 0; q < 4; ++q) {
            v4i_ d4 = dp[q];
            #pragma unroll
            for (int e = 0; e < 4; ++e)
                a[q * 4 + e] = (d4[e] > 0) ? (_Float16)1.0f : (_Float16)0.0f;
        }
        v16h ones;
        #pragma unroll
        for (int e = 0; e < 16; ++e) ones[e] = (_Float16)1.0f;
        v8f c = {};
        c = __builtin_amdgcn_wmma_f32_16x16x32_f16(false, a, false, ones,
                                                   (short)0, c, false, false);
        float s = 0.0f;
        #pragma unroll
        for (int e = 0; e < 8; ++e) s += c[e];
        s += __shfl_xor(s, 16, 32);   // add complementary half-column
        if (lane == 0) s_wsum[wave] = s;
    }
    __syncthreads();
    if (tid == 0) {
        float tot = 0.0f;
        for (int w = 0; w < 32; ++w) tot += s_wsum[w];
        int num_pos = (int)(tot + 0.5f);
        int nn = ratio_p[0] * num_pos;
        if (nn > NA - 1) nn = NA - 1;
        *s_numneg = nn;
    }

    // ---- (b) build keys -----------------------------------------------
    for (int i = tid; i < NA; i += nthr) {
        float v = maxce[(size_t)b * NA + i];
        uint32_t u = __float_as_uint(v);
        uint32_t ordAsc  = (u & 0x80000000u) ? ~u : (u | 0x80000000u);
        uint32_t ordDesc = ~ordAsc;  // ascending sort => descending max_ce
        s_key[i] = ((uint64_t)ordDesc << 32) | (uint32_t)i;
    }
    __syncthreads();

    // bitonic sort ascending, 16384 keys, 1024 threads
    for (int k = 2; k <= NA; k <<= 1) {
        for (int j = k >> 1; j > 0; j >>= 1) {
            for (int i = tid; i < NA; i += nthr) {
                int ixj = i ^ j;
                if (ixj > i) {
                    uint64_t x = s_key[i], y = s_key[ixj];
                    bool up = ((i & k) == 0);
                    if ((x > y) == up) { s_key[i] = y; s_key[ixj] = x; }
                }
            }
            __syncthreads();
        }
    }

    // ---- (c) scatter neg flags ----------------------------------------
    const int nn = *s_numneg;
    for (int p = tid; p < NA; p += nthr) {
        uint32_t a_idx = (uint32_t)(s_key[p] & 0xFFFFFFFFu);
        negmask[(size_t)b * NA + a_idx] = (p < nn) ? (unsigned char)1 : (unsigned char)0;
    }
}

// ---------------------------------------------------------------------------
// Pass 3: out = ce * ((depth > 0) | neg). Wave per row, recompute ce.
// ---------------------------------------------------------------------------
__global__ void k_apply(const float* __restrict__ pred,
                        const float* __restrict__ tgt,
                        const int*   __restrict__ depth,
                        const unsigned char* __restrict__ negmask,
                        float*       __restrict__ out) {
    const int wavesPerBlk = blockDim.x >> 5;
    const int row  = blockIdx.x * wavesPerBlk + (threadIdx.x >> 5);
    const int lane = threadIdx.x & 31;
    if (row >= NB * NA) return;
    const float fm = ((depth[row] > 0) || (negmask[row] != 0)) ? 1.0f : 0.0f;
    const size_t base = (size_t)row * NC;
    for (int c = lane; c < NC; c += 32) {
        float x = pred[base + c];
        float t = tgt[base + c];
        out[base + c] = bce_logits(x, t) * fm;
    }
}

extern "C" void kernel_launch(void* const* d_in, const int* in_sizes, int n_in,
                              void* d_out, int out_size, void* d_ws, size_t ws_size,
                              hipStream_t stream) {
    const float* pred  = (const float*)d_in[0];
    const float* tgt   = (const float*)d_in[1];
    const int*   depth = (const int*)d_in[2];
    const int*   ratio = (const int*)d_in[3];
    float* out = (float*)d_out;

    float* maxce = (float*)d_ws;                                   // 2 MB
    unsigned char* negmask =
        (unsigned char*)d_ws + (size_t)NB * NA * sizeof(float);    // 512 KB

    const int rows = NB * NA;                 // 524288 anchor rows
    dim3 blk(256);                            // 8 waves per block (wave32)
    dim3 grd(rows / 8);                       // 65536 blocks

    k_rowmax<<<grd, blk, 0, stream>>>(pred, tgt, depth, maxce);

    const size_t shmem = (size_t)NA * 8 + 256;   // 128 KB keys + scratch
    k_rank<<<dim3(NB), dim3(1024), shmem, stream>>>(depth, ratio, maxce, negmask);

    k_apply<<<grd, blk, 0, stream>>>(pred, tgt, depth, negmask, out);
}
